// Model_20014547599835
// MI455X (gfx1250) — matrix operand order repaired and checked by reading, not compiled, not run on hardware
//
#include <hip/hip_runtime.h>
#include <hip/hip_bf16.h>
#include <cstdint>

// ---------------------------------------------------------------------------
// MI455X (gfx1250) implementation notes:
//  * Recurrent GEMMs (fused [x|h] @ [W;U]) run on v_wmma_f32_16x16x32_bf16,
//    f32 accumulate, f32 cell state. Weights are L1/L2-resident (<2.6 MB).
//  * Each of 16 WGs owns 16 batch rows and runs the ENTIRE time loop in one
//    kernel: h kept bf16 inside the LDS A-slab, c / h_f32 in LDS, 2 barriers
//    per step, zero launch overhead on the 564-step critical path.
//  * Gate-aligned weight columns (one 256-wide block per gate) make gate math
//    register-resident: wave w owns N-tiles {w,16+w,32+w,48+w}.
//  * wmma_quad: one A-fragment LDS load per K-step feeds 4 WMMAs.
//  * asm memory clobber per time step stops LICM from hoisting (and then
//    spilling) the loop-invariant weight fragments.
// ---------------------------------------------------------------------------

typedef __bf16 bf16_t;
typedef __attribute__((ext_vector_type(16))) __bf16 v16bf;
typedef __attribute__((ext_vector_type(8)))  __bf16 v8bf;
typedef __attribute__((ext_vector_type(8)))  float   v8f;

#define DEV_INLINE __device__ __attribute__((always_inline)) inline

// ---- problem dims ---------------------------------------------------------
constexpr int B_    = 256;
constexpr int T_    = 250;
constexpr int F_    = 14;
constexpr int CL_   = 64;
constexpr int EMB_  = 150;
constexpr int EMBP  = 160;             // EMB padded to mult-of-32
constexpr int GRUU  = 200;
constexpr int GRUP  = 224;             // GRU hidden padded to mult-of-32
constexpr int GRUK  = EMBP + GRUP;     // 384  (concat [x|h] K)
constexpr int GRUNP = 1024;            // gate-aligned: [z|r|x_h|r_h] x 256
constexpr int RNNU  = 256;
constexpr int HID_  = 264;
constexpr int NCLS_ = 8;
constexpr int FXP   = 32;              // feature width 14 padded to 32
constexpr int K1    = FXP + RNNU;      // 288
constexpr int K2    = RNNU + RNNU;     // 512
constexpr int NL    = 1024;            // 4*RNNU

// ---- workspace layout (bytes) --------------------------------------------
constexpr size_t alignup(size_t x) { return (x + 255) & ~size_t(255); }
constexpr size_t OFF_HGRUF = 0;                                    // [B][GRUP] f32
constexpr size_t OFF_H2F   = alignup(OFF_HGRUF + (size_t)B_ * GRUP * 4);
constexpr size_t OFF_WGRU  = alignup(OFF_H2F   + (size_t)B_ * RNNU * 4);
constexpr size_t OFF_W1    = alignup(OFF_WGRU  + (size_t)GRUK * GRUNP * 2);
constexpr size_t OFF_W2    = alignup(OFF_W1    + (size_t)K1 * NL * 2);
constexpr size_t OFF_EMB   = alignup(OFF_W2    + (size_t)K2 * NL * 2);
constexpr size_t OFF_FEAT  = alignup(OFF_EMB   + (size_t)B_ * CL_ * EMBP * 2);
constexpr size_t OFF_SEQ   = alignup(OFF_FEAT  + (size_t)B_ * T_ * FXP * 2);
constexpr size_t OFF_CHARH = alignup(OFF_SEQ   + (size_t)T_ * B_ * RNNU * 2);
constexpr size_t OFF_CHARO = alignup(OFF_CHARH + (size_t)B_ * HID_ * 4);
constexpr size_t OFF_FEATO = alignup(OFF_CHARO + (size_t)B_ * HID_ * 4);
constexpr size_t OFF_HEADH = alignup(OFF_FEATO + (size_t)B_ * HID_ * 4);

// ---- helpers --------------------------------------------------------------
DEV_INLINE float sigmoidf_(float x) { return 1.f / (1.f + __expf(-x)); }

// Quad-tile GEMM: acc[g] += A(16xKTOT, LDS) * B(:, g*256 + nbase .. +16).
// One A-fragment load per K-step feeds all four gate tiles.
template <int KTOT, int LDW>
DEV_INLINE void wmma_quad(const bf16_t* slab, const bf16_t* __restrict__ W,
                          int nbase, v8f acc[4])
{
    const int lane = threadIdx.x & 31;
    const int mr   = lane & 15;
    const int hsel = lane >> 4;
    const bf16_t* arow = slab + mr * KTOT;
    const bf16_t* brow = W + (size_t)(8 * hsel) * LDW + nbase + mr;
#pragma unroll
    for (int k0 = 0; k0 < KTOT; k0 += 32) {
        // A fragment (16-bit 16x32 layout): lane holds row mr, K pattern
        // {k0+8h..+7} U {k0+16+8h..+7} -> two aligned 16B LDS loads.
        v8bf lo = *(const v8bf*)(arow + k0 + 8 * hsel);
        v8bf hi = *(const v8bf*)(arow + k0 + 16 + 8 * hsel);
        union { v16bf v; v8bf h[2]; } a;
        a.h[0] = lo;
        a.h[1] = hi;
        const bf16_t* bk = brow + (size_t)k0 * LDW;
#pragma unroll
        for (int g = 0; g < 4; ++g) {
            // B fragment (32x16): lane holds col mr, K pattern matching A.
            v16bf bfrag;
#pragma unroll
            for (int s = 0; s < 16; ++s)
                bfrag[s] = bk[(size_t)(s < 8 ? s : s + 8) * LDW + g * 256];
            acc[g] = __builtin_amdgcn_wmma_f32_16x16x32_bf16(
                false, a.v, false, bfrag, (short)0, acc[g], false, false);
        }
    }
}

// ---- prep kernels ---------------------------------------------------------
// GRU W_cat, gate-aligned columns: n = g*256 + u, g in {z, r, x_h, r_h}.
// Rows [0,150)=gru_W, [160,360)=gru_U, rest zero.  z,r columns carry W and U
// summed; x_h carries W only; r_h carries U only (reset_after semantics).
__global__ void prep_gru_w_kernel(const float* __restrict__ W,
                                  const float* __restrict__ U,
                                  bf16_t* __restrict__ out)
{
    int i = blockIdx.x * blockDim.x + threadIdx.x;
    if (i >= GRUK * GRUNP) return;
    int k = i / GRUNP, n = i - k * GRUNP;
    int g = n >> 8, u = n & 255;
    float v = 0.f;
    if (u < GRUU) {
        const bool kw = (k < EMB_);
        const bool ku = (k >= EMBP && k < EMBP + GRUU);
        if (g == 0) {                    // z: fused x+h
            if (kw) v = W[k * 600 + u];
            else if (ku) v = U[(k - EMBP) * 600 + u];
        } else if (g == 1) {             // r: fused x+h
            if (kw) v = W[k * 600 + 200 + u];
            else if (ku) v = U[(k - EMBP) * 600 + 200 + u];
        } else if (g == 2) {             // x_h (W only)
            if (kw) v = W[k * 600 + 400 + u];
        } else {                         // r_h (U only)
            if (ku) v = U[(k - EMBP) * 600 + 400 + u];
        }
    }
    out[i] = (bf16_t)v;
}

// LSTM W_cat: rows [0,kwreal)=W, rows [KX,KX+256)=U, rest zero.
__global__ void prep_lstm_w_kernel(const float* __restrict__ W, int kwreal,
                                   int KX, const float* __restrict__ U,
                                   bf16_t* __restrict__ out, int KTOT)
{
    int i = blockIdx.x * blockDim.x + threadIdx.x;
    if (i >= KTOT * NL) return;
    int k = i / NL, n = i - k * NL;
    float v = 0.f;
    if (k < kwreal)   v = W[k * NL + n];
    else if (k >= KX) v = U[(k - KX) * NL + n];
    out[i] = (bf16_t)v;
}

__global__ void prep_emb_kernel(const int* __restrict__ chars,
                                const float* __restrict__ table,
                                bf16_t* __restrict__ out)
{
    int i = blockIdx.x * blockDim.x + threadIdx.x;
    if (i >= B_ * CL_ * EMBP) return;
    int b = i / (CL_ * EMBP);
    int r = i - b * (CL_ * EMBP);
    int t = r / EMBP, k = r - t * EMBP;
    int idx = chars[b * CL_ + t];
    float v = (k < EMB_) ? table[(size_t)idx * EMB_ + k] : 0.f;
    out[i] = (bf16_t)v;
}

__global__ void prep_feat_kernel(const float* __restrict__ fin,
                                 bf16_t* __restrict__ out)
{
    int i = blockIdx.x * blockDim.x + threadIdx.x;
    if (i >= B_ * T_ * FXP) return;
    int b = i / (T_ * FXP);
    int r = i - b * (T_ * FXP);
    int t = r / FXP, k = r - t * FXP;
    float v = (k < F_) ? fin[((size_t)b * T_ + t) * F_ + k] : 0.f;
    out[i] = (bf16_t)v;
}

// ---- GRU: whole 64-step recurrence in one kernel --------------------------
__global__ __launch_bounds__(512)
void gru_seq_kernel(const bf16_t* __restrict__ embbf,
                    const bf16_t* __restrict__ Wcat,
                    const float* __restrict__ grub,
                    float* __restrict__ h_out)
{
    __shared__ __attribute__((aligned(32))) bf16_t slab[16 * GRUK]; // x|h, 12 KB
    __shared__ float hf[16 * GRUP];                                 // f32 h, 14 KB
    const int m0  = blockIdx.x * 16;
    const int tid = threadIdx.x;

    for (int i = tid; i < 16 * GRUP; i += 512) {         // zero h state
        int m = i / GRUP, k = i - m * GRUP;
        slab[m * GRUK + EMBP + k] = (bf16_t)0.f;
        hf[i] = 0.f;
    }

    const int wave = tid >> 5;
    const int lane = tid & 31;
    const int u    = wave * 16 + (lane & 15);

    float bz = 0.f, br = 0.f, bxh = 0.f, brh = 0.f;
    if (u < GRUU) {
        bz  = grub[u]       + grub[600 + u];
        br  = grub[200 + u] + grub[800 + u];
        bxh = grub[400 + u];
        brh = grub[1000 + u];
    }
    const v8f vzero = {0.f, 0.f, 0.f, 0.f, 0.f, 0.f, 0.f, 0.f};

#pragma unroll 1
    for (int t = 0; t < CL_; ++t) {
        // keep weight-fragment loads inside the loop (L1/L2-hot), no spills
        asm volatile("" ::: "memory");
        for (int i = tid; i < 16 * EMBP; i += 512) {     // refresh x slice
            int m = i / EMBP, k = i - m * EMBP;
            slab[m * GRUK + k] = embbf[((size_t)(m0 + m) * CL_ + t) * EMBP + k];
        }
        __syncthreads();
        v8f acc[4] = {vzero, vzero, vzero, vzero};       // z, r, x_h, r_h
        wmma_quad<GRUK, GRUNP>(slab, Wcat, wave * 16, acc);
        __syncthreads();                                 // all reads of h done
        if (u < GRUU) {
#pragma unroll
            for (int v = 0; v < 8; ++v) {
                int m = v + 8 * (lane >> 4);
                float z  = sigmoidf_(acc[0][v] + bz);
                float r  = sigmoidf_(acc[1][v] + br);
                float hh = tanhf(acc[2][v] + bxh + r * (acc[3][v] + brh));
                float hp = hf[m * GRUP + u];
                float hn = z * hp + (1.f - z) * hh;
                hf[m * GRUP + u] = hn;
                slab[m * GRUK + EMBP + u] = (bf16_t)hn;
                if (t == CL_ - 1) h_out[(size_t)(m0 + m) * GRUP + u] = hn;
            }
        }
        // next iteration's x-fill touches disjoint LDS; the barrier after the
        // fill orders these h writes before the next GEMM.
    }
}

// ---- LSTM: whole 250-step recurrence in one kernel ------------------------
template <int KX, int KTOT>
__global__ __launch_bounds__(512)
void lstm_seq_kernel(const bf16_t* __restrict__ xsrc, size_t x_bstride,
                     size_t x_tstride, const bf16_t* __restrict__ Wcat,
                     const float* __restrict__ bias,
                     bf16_t* __restrict__ seq_out,
                     float* __restrict__ h_final, int TSTEPS)
{
    __shared__ __attribute__((aligned(32))) bf16_t slab[16 * KTOT]; // x|h
    __shared__ float cf[16 * RNNU];                                 // f32 cell
    const int m0  = blockIdx.x * 16;
    const int tid = threadIdx.x;

    for (int i = tid; i < 16 * RNNU; i += 512) {         // zero h, c state
        int m = i >> 8, k = i & 255;
        slab[m * KTOT + KX + k] = (bf16_t)0.f;
        cf[i] = 0.f;
    }

    const int wave = tid >> 5;
    const int lane = tid & 31;
    const int u    = wave * 16 + (lane & 15);
    const float bi_ = bias[u];
    const float bf_ = bias[256 + u];
    const float bg_ = bias[512 + u];
    const float bo_ = bias[768 + u];
    const v8f vzero = {0.f, 0.f, 0.f, 0.f, 0.f, 0.f, 0.f, 0.f};

#pragma unroll 1
    for (int t = 0; t < TSTEPS; ++t) {
        // keep weight-fragment loads inside the loop (L1/L2-hot), no spills
        asm volatile("" ::: "memory");
        for (int i = tid; i < 16 * KX; i += 512) {       // refresh x slice
            int m = i / KX, k = i - m * KX;
            slab[m * KTOT + k] =
                xsrc[(size_t)(m0 + m) * x_bstride + (size_t)t * x_tstride + k];
        }
        __syncthreads();
        // Gate-aligned tiles: i/f/g/o for units [16w,16w+16) in matching slots.
        v8f acc[4] = {vzero, vzero, vzero, vzero};       // i, f, g, o
        wmma_quad<KTOT, NL>(slab, Wcat, wave * 16, acc);
        __syncthreads();                                 // all reads of h done
#pragma unroll
        for (int v = 0; v < 8; ++v) {
            int m = v + 8 * (lane >> 4);
            float cp = cf[m * RNNU + u];
            float cn = sigmoidf_(acc[1][v] + bf_) * cp
                     + sigmoidf_(acc[0][v] + bi_) * tanhf(acc[2][v] + bg_);
            float hn = sigmoidf_(acc[3][v] + bo_) * tanhf(cn);
            cf[m * RNNU + u] = cn;
            slab[m * KTOT + KX + u] = (bf16_t)hn;
            if (seq_out)
                seq_out[(size_t)t * B_ * RNNU + (size_t)(m0 + m) * RNNU + u] =
                    (bf16_t)hn;
            if (h_final && t == TSTEPS - 1)
                h_final[(size_t)(m0 + m) * RNNU + u] = hn;
        }
    }
}

// ---- tiny head layers (exact f32) ----------------------------------------
__global__ void dense_kernel(const float* __restrict__ X, int ldx, int K,
                             const float* __restrict__ W,
                             const float* __restrict__ bias,
                             float* __restrict__ Y, int N, int relu)
{
    int i = blockIdx.x * blockDim.x + threadIdx.x;
    if (i >= B_ * N) return;
    int b = i / N, n = i - b * N;
    float s = bias[n];
    for (int k = 0; k < K; ++k) s = fmaf(X[b * ldx + k], W[k * N + n], s);
    if (relu) s = fmaxf(s, 0.f);
    Y[b * N + n] = s;
}

__global__ void dense2_kernel(const float* __restrict__ X1,
                              const float* __restrict__ X2, int K1e, int K2e,
                              const float* __restrict__ W,
                              const float* __restrict__ bias,
                              float* __restrict__ Y, int N, int relu)
{
    int i = blockIdx.x * blockDim.x + threadIdx.x;
    if (i >= B_ * N) return;
    int b = i / N, n = i - b * N;
    float s = bias[n];
    for (int k = 0; k < K1e; ++k) s = fmaf(X1[b * K1e + k], W[k * N + n], s);
    for (int k = 0; k < K2e; ++k) s = fmaf(X2[b * K2e + k], W[(K1e + k) * N + n], s);
    if (relu) s = fmaxf(s, 0.f);
    Y[b * N + n] = s;
}

__global__ void softmax_kernel(float* Y, int N)
{
    int b = blockIdx.x * blockDim.x + threadIdx.x;
    if (b >= B_) return;
    float mx = -3.4e38f;
    for (int n = 0; n < N; ++n) mx = fmaxf(mx, Y[b * N + n]);
    float s = 0.f;
    for (int n = 0; n < N; ++n) {
        float e = __expf(Y[b * N + n] - mx);
        Y[b * N + n] = e;
        s += e;
    }
    float inv = 1.f / s;
    for (int n = 0; n < N; ++n) Y[b * N + n] *= inv;
}

// ---------------------------------------------------------------------------
static inline int cdiv(int a, int b) { return (a + b - 1) / b; }

extern "C" void kernel_launch(void* const* d_in, const int* in_sizes, int n_in,
                              void* d_out, int out_size, void* d_ws,
                              size_t ws_size, hipStream_t stream)
{
    (void)in_sizes; (void)n_in; (void)out_size; (void)ws_size;

    const float* feature_inputs = (const float*)d_in[1];
    const int*   char_inputs    = (const int*)d_in[2];
    const float* char_emb       = (const float*)d_in[3];
    const float* gru_W = (const float*)d_in[4];
    const float* gru_U = (const float*)d_in[5];
    const float* gru_b = (const float*)d_in[6];
    const float* l1_W  = (const float*)d_in[7];
    const float* l1_U  = (const float*)d_in[8];
    const float* l1_b  = (const float*)d_in[9];
    const float* l2_W  = (const float*)d_in[10];
    const float* l2_U  = (const float*)d_in[11];
    const float* l2_b  = (const float*)d_in[12];
    const float* cd1_W = (const float*)d_in[13];
    const float* cd1_b = (const float*)d_in[14];
    const float* cd2_W = (const float*)d_in[15];
    const float* cd2_b = (const float*)d_in[16];
    const float* fd1_W = (const float*)d_in[17];
    const float* fd1_b = (const float*)d_in[18];
    const float* d1_W  = (const float*)d_in[19];
    const float* d1_b  = (const float*)d_in[20];
    const float* sm_W  = (const float*)d_in[21];
    const float* sm_b  = (const float*)d_in[22];

    char* ws = (char*)d_ws;
    float*  h_gru_f = (float*) (ws + OFF_HGRUF);
    float*  h2_f    = (float*) (ws + OFF_H2F);
    bf16_t* Wg      = (bf16_t*)(ws + OFF_WGRU);
    bf16_t* W1      = (bf16_t*)(ws + OFF_W1);
    bf16_t* W2      = (bf16_t*)(ws + OFF_W2);
    bf16_t* embbf   = (bf16_t*)(ws + OFF_EMB);
    bf16_t* featbf  = (bf16_t*)(ws + OFF_FEAT);
    bf16_t* seqbf   = (bf16_t*)(ws + OFF_SEQ);
    float*  charH   = (float*) (ws + OFF_CHARH);
    float*  charO   = (float*) (ws + OFF_CHARO);
    float*  featO   = (float*) (ws + OFF_FEATO);
    float*  headH   = (float*) (ws + OFF_HEADH);
    float*  out     = (float*)d_out;

    // weight / activation prep (f32 -> padded bf16)
    prep_gru_w_kernel<<<cdiv(GRUK * GRUNP, 256), 256, 0, stream>>>(gru_W, gru_U, Wg);
    prep_lstm_w_kernel<<<cdiv(K1 * NL, 256), 256, 0, stream>>>(l1_W, F_,   FXP,  l1_U, W1, K1);
    prep_lstm_w_kernel<<<cdiv(K2 * NL, 256), 256, 0, stream>>>(l2_W, RNNU, RNNU, l2_U, W2, K2);
    prep_emb_kernel<<<cdiv(B_ * CL_ * EMBP, 256), 256, 0, stream>>>(char_inputs, char_emb, embbf);
    prep_feat_kernel<<<cdiv(B_ * T_ * FXP, 256), 256, 0, stream>>>(feature_inputs, featbf);

    // full recurrences, one kernel each (state LDS-resident)
    gru_seq_kernel<<<16, 512, 0, stream>>>(embbf, Wg, gru_b, h_gru_f);
    lstm_seq_kernel<FXP, K1><<<16, 512, 0, stream>>>(
        featbf, (size_t)(T_ * FXP), (size_t)FXP, W1, l1_b, seqbf, nullptr, T_);
    lstm_seq_kernel<RNNU, K2><<<16, 512, 0, stream>>>(
        seqbf, (size_t)RNNU, (size_t)(B_ * RNNU), W2, l2_b, nullptr, h2_f, T_);

    // head (tiny, exact f32)
    dense_kernel<<<cdiv(B_ * HID_, 256), 256, 0, stream>>>(h_gru_f, GRUP, GRUU, cd1_W, cd1_b, charH, HID_, 1);
    dense_kernel<<<cdiv(B_ * HID_, 256), 256, 0, stream>>>(charH, HID_, HID_, cd2_W, cd2_b, charO, HID_, 0);
    softmax_kernel<<<1, 256, 0, stream>>>(charO, HID_);
    dense_kernel<<<cdiv(B_ * HID_, 256), 256, 0, stream>>>(h2_f, RNNU, RNNU, fd1_W, fd1_b, featO, HID_, 1);
    dense2_kernel<<<cdiv(B_ * HID_, 256), 256, 0, stream>>>(featO, charO, HID_, HID_, d1_W, d1_b, headH, HID_, 1);
    dense_kernel<<<cdiv(B_ * NCLS_, 256), 256, 0, stream>>>(headH, HID_, HID_, sm_W, sm_b, out, NCLS_, 0);
    softmax_kernel<<<1, 256, 0, stream>>>(out, NCLS_);
}
